// MambaTrajEncoder_46935402611373
// MI455X (gfx1250) — matrix operand order repaired and checked by reading, not compiled, not run on hardware
//
#include <hip/hip_runtime.h>

// ---------------- model dims ----------------
#define NLAY 3
#define DM   512
#define DI   1024
#define DS   16
#define DC   4
#define DR   32
#define BB   4
#define LL   2048
#define TDIM 128
#define MROWS (BB*LL)   // 8192 token rows

typedef __attribute__((ext_vector_type(16))) _Float16 v16h;
typedef __attribute__((ext_vector_type(8)))  _Float16 v8h;
typedef __attribute__((ext_vector_type(8)))  float    v8f;

// ---------------- CDNA5 async global->LDS copy (guarded) ----------------
#if defined(__has_builtin)
# if __has_builtin(__builtin_amdgcn_global_load_async_to_lds_b128)
#  define HAVE_ASYNC_LDS 1
# endif
#endif
#ifndef HAVE_ASYNC_LDS
# define HAVE_ASYNC_LDS 0
#endif

#if HAVE_ASYNC_LDS
# if __has_builtin(__builtin_amdgcn_s_wait_asynccnt)
#  define ASYNC_WAIT_N(n) __builtin_amdgcn_s_wait_asynccnt(n)
# else
#  define ASYNC_WAIT_N(n) asm volatile("s_wait_asynccnt %0" :: "i"(n) : "memory")
# endif
// Builtin prototype (from hipcc diagnostic): pointers to 128-bit int vectors,
// global side AS1, LDS side AS3.  Flat-address low 32 bits == LDS byte offset
// (ISA aperture mapping), so AS pointers are materialized via integer casts.
typedef int b128_t __attribute__((vector_size(16)));
__device__ __forceinline__ void async_cp16(const _Float16* g, _Float16* l)
{
    __builtin_amdgcn_global_load_async_to_lds_b128(
        (__attribute__((address_space(1))) b128_t*)(unsigned long long)g,
        (__attribute__((address_space(3))) b128_t*)(unsigned int)(unsigned long long)l,
        0, 0);
}
#else
# define ASYNC_WAIT_N(n) ((void)0)
#endif

// ---------------------------------------------------------------------------
// f32 -> f16 weight pre-conversion (once per launch)
// ---------------------------------------------------------------------------
__global__ void f32_to_f16_kernel(const float* __restrict__ src,
                                  _Float16* __restrict__ dst, int n)
{
    int i = blockIdx.x * blockDim.x + threadIdx.x;
    if (i < n) dst[i] = (_Float16)src[i];
}

// ---------------------------------------------------------------------------
// LayerNorm over DM=512: one wave per row; writes f16 (per-layer) or f32 (final)
// ---------------------------------------------------------------------------
__global__ void ln_kernel(const float* __restrict__ h,
                          const float* __restrict__ g,
                          const float* __restrict__ bta,
                          _Float16* __restrict__ out16,
                          float* __restrict__ out32, int rows)
{
    int gt   = blockIdx.x * blockDim.x + threadIdx.x;
    int row  = gt >> 5;
    int lane = gt & 31;
    if (row >= rows) return;
    const float* x = h + (size_t)row * DM;
    float v[16];
    float s = 0.f, s2 = 0.f;
#pragma unroll
    for (int j = 0; j < 16; j++) {
        float t = x[lane + 32 * j];
        v[j] = t; s += t; s2 += t * t;
    }
#pragma unroll
    for (int off = 16; off >= 1; off >>= 1) {
        s  += __shfl_xor(s,  off, 32);
        s2 += __shfl_xor(s2, off, 32);
    }
    float mu  = s  * (1.f / DM);
    float var = s2 * (1.f / DM) - mu * mu;
    float inv = rsqrtf(var + 1e-5f);
#pragma unroll
    for (int j = 0; j < 16; j++) {
        int c = lane + 32 * j;
        float o = (v[j] - mu) * inv * g[c] + bta[c];
        if (out16) out16[(size_t)row * DM + c] = (_Float16)o;
        else       out32[(size_t)row * DM + c] = o;
    }
}

// ---------------------------------------------------------------------------
// Causal depthwise conv (DC=4) + bias + SiLU
// ---------------------------------------------------------------------------
__global__ void conv_silu_kernel(const _Float16* __restrict__ xz,
                                 const float* __restrict__ cw,
                                 const float* __restrict__ cb,
                                 _Float16* __restrict__ xc16,
                                 float* __restrict__ xc32)
{
    int idx = blockIdx.x * blockDim.x + threadIdx.x;   // [0, MROWS*DI)
    int d  = idx & (DI - 1);
    int bl = idx >> 10;
    int l  = bl & (LL - 1);
    float acc = cb[d];
#pragma unroll
    for (int j = 0; j < DC; j++) {
        int tl = l - (DC - 1) + j;
        if (tl >= 0)
            acc += (float)xz[(size_t)(bl - (DC - 1) + j) * (2 * DI) + d] * cw[d * DC + j];
    }
    float sg = 1.f / (1.f + __expf(-acc));
    float o  = acc * sg;
    xc16[(size_t)bl * DI + d] = (_Float16)o;
    xc32[(size_t)bl * DI + d] = o;
}

// ---------------------------------------------------------------------------
// Selective scan: one thread per (b,d), DS=16 states in registers; fuses
// y = (scan + D*x) * silu(z), stores f16 for out-projection.
// ---------------------------------------------------------------------------
__global__ void scan_kernel(const float* __restrict__ dt,
                            const float* __restrict__ xc,
                            const _Float16* __restrict__ xz,
                            const float* __restrict__ xdbl,
                            const float* __restrict__ Alog,
                            const float* __restrict__ Dp,
                            _Float16* __restrict__ y16)
{
    int gid = blockIdx.x * blockDim.x + threadIdx.x;   // [0, BB*DI)
    int d = gid & (DI - 1);
    int b = gid >> 10;
    float Ad[DS];
#pragma unroll
    for (int s = 0; s < DS; s++) Ad[s] = -__expf(Alog[d * DS + s]);
    float Dv = Dp[d];
    float hs[DS];
#pragma unroll
    for (int s = 0; s < DS; s++) hs[s] = 0.f;

    for (int t = 0; t < LL; t++) {
        size_t r   = (size_t)b * LL + t;
        float dtv  = dt[r * DI + d];
        float xv   = xc[r * DI + d];
        const float4* bc = (const float4*)(xdbl + r * 64 + DR);
        float Bv[DS], Cv[DS];
#pragma unroll
        for (int q = 0; q < 4; q++) {
            float4 vB = bc[q];
            float4 vC = bc[4 + q];
            Bv[4*q+0] = vB.x; Bv[4*q+1] = vB.y; Bv[4*q+2] = vB.z; Bv[4*q+3] = vB.w;
            Cv[4*q+0] = vC.x; Cv[4*q+1] = vC.y; Cv[4*q+2] = vC.z; Cv[4*q+3] = vC.w;
        }
        float dx = dtv * xv;
        float y  = 0.f;
#pragma unroll
        for (int s = 0; s < DS; s++) {
            float dA = __expf(dtv * Ad[s]);
            hs[s] = dA * hs[s] + dx * Bv[s];
            y += hs[s] * Cv[s];
        }
        y += Dv * xv;
        float zv = (float)xz[r * (2 * DI) + DI + d];
        float sg = 1.f / (1.f + __expf(-zv));
        y *= zv * sg;
        y16[r * DI + d] = (_Float16)y;
    }
}

// ---------------------------------------------------------------------------
// WMMA GEMM:  C[M,N] = epi( A[M,K] @ W[N,K]^T + bias ),  W in f16.
// Block 256 thr = 8 waves, block tile 128x128, wave tile 64x32, K-step 32.
// 2-stage software pipeline: double-buffered LDS tiles; tile s+1's
// GLOBAL_LOAD_ASYNC_TO_LDS_B128 copies stay in flight across tile s's WMMA
// burst; s_wait_asynccnt <= ops(s+1) guarantees tile s complete.
// EPI: 0=store, 1=+bias, 2=+bias+softplus, 3=residual add (C += acc)
// ---------------------------------------------------------------------------
#define LDSW  40          // padded LDS row stride (halves): 16B aligned
#define TILEH (128*LDSW)  // halves per tile buffer

__device__ __forceinline__ v16h frag_a(const _Float16* s, int rowBase)
{
    int lane = threadIdx.x & 31;
    const _Float16* p = s + (rowBase + (lane & 15)) * LDSW + ((lane >> 4) * 8);
    v8h lo = *(const v8h*)p;
    v8h hi = *(const v8h*)(p + 16);
    v16h f;
#pragma unroll
    for (int i = 0; i < 8; i++) { f[i] = lo[i]; f[i + 8] = hi[i]; }
    return f;
}

__device__ __forceinline__ v16h frag_b(const _Float16* s, int rowBase)
{
    int lane = threadIdx.x & 31;
    const _Float16* p = s + (rowBase + (lane & 15)) * LDSW + ((lane >> 4) * 8);
    v8h lo = *(const v8h*)p;
    v8h hi = *(const v8h*)(p + 16);
    v16h f;
#pragma unroll
    for (int i = 0; i < 8; i++) { f[i] = lo[i]; f[i + 8] = hi[i]; }
    return f;
}

template <typename AT, typename OT, int EPI>
__global__ void gemm_wmma_kernel(const AT* __restrict__ A, int lda,
                                 const _Float16* __restrict__ W,
                                 const float* __restrict__ bias,
                                 OT* __restrict__ C, int N, int K)
{
    __shared__ _Float16 sA[2 * TILEH];
    __shared__ _Float16 sW[2 * TILEH];

    const int mb    = blockIdx.x * 128;
    const int nb    = blockIdx.y * 128;
    const int wave  = threadIdx.x >> 5;
    const int lane  = threadIdx.x & 31;
    const int waveM = wave >> 2;          // 0..1
    const int waveN = wave & 3;           // 0..3
    const int tr    = threadIdx.x >> 1;   // 0..127 staging row
    const int seg   = threadIdx.x & 1;    // 16-half segment

    // clamp W row for tiles overhanging N (e.g. N=64 x-proj): loads valid dummy
    // data, keeps EXEC + per-wave ASYNCcnt uniform; epilogue discards col >= N.
    int wrow = nb + tr;
    if (wrow >= N) wrow = N - 1;
    const _Float16* wsrc0 = W + (size_t)wrow * K + seg * 16;
    // per-wave async ops per tile: 2 (W) + 2 more if A is f16
    constexpr int PIPE_OPS = (sizeof(AT) == 2) ? 4 : 2;

    // tile staging: global -> LDS buffer `buf`
    auto stage = [&](int step, int buf) {
        int kt = step << 5;
        // ---- A tile ----
        _Float16* dstA = sA + buf * TILEH + tr * LDSW + seg * 16;
        const AT* srcA = A + (size_t)(mb + tr) * lda + kt + seg * 16;
        if (sizeof(AT) == 4) {            // f32 -> f16 convert path (embed / dt)
            __builtin_prefetch(srcA + 32, 0, 0);   // next K tile
            const float4* s4 = (const float4*)(const void*)srcA;
#pragma unroll
            for (int q = 0; q < 4; q++) {
                float4 v = s4[q];
                dstA[4*q+0] = (_Float16)v.x; dstA[4*q+1] = (_Float16)v.y;
                dstA[4*q+2] = (_Float16)v.z; dstA[4*q+3] = (_Float16)v.w;
            }
        } else {
#if HAVE_ASYNC_LDS
            async_cp16((const _Float16*)(const void*)srcA,     dstA);
            async_cp16((const _Float16*)(const void*)srcA + 8, dstA + 8);
#else
            *(v8h*)dstA       = *(const v8h*)(const void*)srcA;
            *(v8h*)(dstA + 8) = *(const v8h*)(const void*)(srcA + 8);
#endif
        }
        // ---- W tile (f16) ----
        _Float16* dstW = sW + buf * TILEH + tr * LDSW + seg * 16;
        const _Float16* srcW = wsrc0 + kt;
#if HAVE_ASYNC_LDS
        async_cp16(srcW,     dstW);
        async_cp16(srcW + 8, dstW + 8);
#else
        *(v8h*)dstW       = *(const v8h*)srcW;
        *(v8h*)(dstW + 8) = *(const v8h*)(srcW + 8);
#endif
    };

    v8f zero = {};
    v8f acc[4][2];
#pragma unroll
    for (int a = 0; a < 4; a++)
#pragma unroll
        for (int b2 = 0; b2 < 2; b2++) acc[a][b2] = zero;

    const int nsteps = K >> 5;
    stage(0, 0);                          // prologue

    for (int s = 0; s < nsteps; s++) {
        const int cur = s & 1;
        if (s + 1 < nsteps) {
            stage(s + 1, cur ^ 1);        // overlap next fetch with this compute
            ASYNC_WAIT_N(PIPE_OPS);       // tile s done; tile s+1 in flight
        } else {
            ASYNC_WAIT_N(0);
        }
        __syncthreads();                  // all waves' tile-s copies visible

        const _Float16* cA = sA + cur * TILEH;
        const _Float16* cW = sW + cur * TILEH;
        v16h af[4], bf[2];
#pragma unroll
        for (int fm = 0; fm < 4; fm++) af[fm] = frag_a(cA, waveM * 64 + fm * 16);
#pragma unroll
        for (int fn = 0; fn < 2; fn++) bf[fn] = frag_b(cW, waveN * 32 + fn * 16);
#pragma unroll
        for (int fm = 0; fm < 4; fm++)
#pragma unroll
            for (int fn = 0; fn < 2; fn++)
                acc[fm][fn] = __builtin_amdgcn_wmma_f32_16x16x32_f16(
                    false, af[fm], false, bf[fn], (short)0, acc[fm][fn], false, false);

        __syncthreads();                  // reads done before buf reuse at s+2
    }

    // epilogue: lane&15 = col, (lane>>4)*8 + r = fragment row
    const int colw = lane & 15;
    const int rsel = (lane >> 4) * 8;
#pragma unroll
    for (int fn = 0; fn < 2; fn++) {
        int col = nb + waveN * 32 + fn * 16 + colw;
        if (col >= N) continue;
        float bv = (EPI == 1 || EPI == 2) ? bias[col] : 0.f;
#pragma unroll
        for (int fm = 0; fm < 4; fm++) {
#pragma unroll
            for (int r = 0; r < 8; r++) {
                int row = mb + waveM * 64 + fm * 16 + rsel + r;
                float v = acc[fm][fn][r] + bv;
                if (EPI == 2) v = (v > 20.f) ? v : log1pf(__expf(v));
                size_t idx = (size_t)row * N + col;
                if (EPI == 3) C[idx] += (OT)v;
                else          C[idx]  = (OT)v;
            }
        }
    }
}

// ---------------------------------------------------------------------------
extern "C" void kernel_launch(void* const* d_in, const int* in_sizes, int n_in,
                              void* d_out, int out_size, void* d_ws, size_t ws_size,
                              hipStream_t stream)
{
    (void)in_sizes; (void)n_in; (void)out_size; (void)ws_size;

    const float* seq    = (const float*)d_in[0];
    const float* w_inp  = (const float*)d_in[1];
    const float* b_inp  = (const float*)d_in[2];
    const float* ln_g   = (const float*)d_in[3];
    const float* ln_b   = (const float*)d_in[4];
    const float* w_inpj = (const float*)d_in[5];
    const float* conv_w = (const float*)d_in[6];
    const float* conv_b = (const float*)d_in[7];
    const float* w_xprj = (const float*)d_in[8];
    const float* w_dt   = (const float*)d_in[9];
    const float* b_dt   = (const float*)d_in[10];
    const float* A_log  = (const float*)d_in[11];
    const float* Dp     = (const float*)d_in[12];
    const float* w_out  = (const float*)d_in[13];
    const float* out_g  = (const float*)d_in[14];
    const float* out_b  = (const float*)d_in[15];

    char* ws = (char*)d_ws;
    size_t off = 0;
    auto carve = [&](size_t bytes) -> char* {
        char* p = ws + off;
        off += (bytes + 255) & ~(size_t)255;
        return p;
    };
    float*    h    = (float*)    carve((size_t)MROWS * DM * 4);
    _Float16* xh   = (_Float16*) carve((size_t)MROWS * DM * 2);
    _Float16* xz   = (_Float16*) carve((size_t)MROWS * 2 * DI * 2);
    _Float16* xc16 = (_Float16*) carve((size_t)MROWS * DI * 2);
    float*    xc32 = (float*)    carve((size_t)MROWS * DI * 4);
    float*    xdbl = (float*)    carve((size_t)MROWS * 64 * 4);
    float*    dtb  = (float*)    carve((size_t)MROWS * DI * 4);
    _Float16* y16  = (_Float16*) carve((size_t)MROWS * DI * 2);
    // f16 weight copies
    _Float16* w_inp16  = (_Float16*) carve((size_t)DM * TDIM * 2);
    _Float16* w_inpj16 = (_Float16*) carve((size_t)NLAY * 2 * DI * DM * 2);
    _Float16* w_xprj16 = (_Float16*) carve((size_t)NLAY * 64 * DI * 2);
    _Float16* w_dt16   = (_Float16*) carve((size_t)NLAY * DI * DR * 2);
    _Float16* w_out16  = (_Float16*) carve((size_t)NLAY * DM * DI * 2);

    dim3 blk(256);

    // weight pre-conversion (f32 -> f16)
    auto cvt = [&](const float* s, _Float16* d, int n) {
        f32_to_f16_kernel<<<dim3((n + 255) / 256), blk, 0, stream>>>(s, d, n);
    };
    cvt(w_inp,  w_inp16,  DM * TDIM);
    cvt(w_inpj, w_inpj16, NLAY * 2 * DI * DM);
    cvt(w_xprj, w_xprj16, NLAY * 64 * DI);
    cvt(w_dt,   w_dt16,   NLAY * DI * DR);
    cvt(w_out,  w_out16,  NLAY * DM * DI);

    // embed: h = seq @ w_inp^T + b_inp
    gemm_wmma_kernel<float, float, 1>
        <<<dim3(MROWS / 128, DM / 128), blk, 0, stream>>>(
            seq, TDIM, w_inp16, b_inp, h, DM, TDIM);

    for (int i = 0; i < NLAY; i++) {
        ln_kernel<<<dim3(MROWS * 32 / 256), blk, 0, stream>>>(
            h, ln_g + i * DM, ln_b + i * DM, xh, nullptr, MROWS);

        gemm_wmma_kernel<_Float16, _Float16, 0>
            <<<dim3(MROWS / 128, (2 * DI) / 128), blk, 0, stream>>>(
                xh, DM, w_inpj16 + (size_t)i * 2 * DI * DM, nullptr, xz, 2 * DI, DM);

        conv_silu_kernel<<<dim3((size_t)MROWS * DI / 256), blk, 0, stream>>>(
            xz, conv_w + (size_t)i * DI * DC, conv_b + (size_t)i * DI, xc16, xc32);

        gemm_wmma_kernel<_Float16, float, 0>
            <<<dim3(MROWS / 128, 1), blk, 0, stream>>>(
                xc16, DI, w_xprj16 + (size_t)i * 64 * DI, nullptr, xdbl, 64, DI);

        gemm_wmma_kernel<float, float, 2>
            <<<dim3(MROWS / 128, DI / 128), blk, 0, stream>>>(
                xdbl, 64, w_dt16 + (size_t)i * DI * DR, b_dt + (size_t)i * DI, dtb, DI, DR);

        scan_kernel<<<dim3(BB * DI / 256), blk, 0, stream>>>(
            dtb, xc32, xz, xdbl, A_log + (size_t)i * DI * DS, Dp + (size_t)i * DI, y16);

        gemm_wmma_kernel<_Float16, float, 3>
            <<<dim3(MROWS / 128, DM / 128), blk, 0, stream>>>(
                y16, DI, w_out16 + (size_t)i * DM * DI, nullptr, h, DM, DI);
    }

    ln_kernel<<<dim3(MROWS * 32 / 256), blk, 0, stream>>>(
        h, out_g, out_b, nullptr, (float*)d_out, MROWS);
}
